// ToMel_70798240907634
// MI455X (gfx1250) — matrix operand order repaired and checked by reading, not compiled, not run on hardware
//
#include <hip/hip_runtime.h>
#include <hip/hip_bf16.h>
#include <math.h>

// ---------------- problem constants ----------------
#define S_IN      661500
#define NFFT      2048
#define HOP       512
#define NMELS     128
#define NBINS     1025                 // 1 + NFFT/2
#define NFRAMES   1292                 // 1 + (S_IN)/HOP (floor)
#define BATCH     8
#define NCH       2

#define FRAME_BLK 64                   // frames per workgroup (4 waves x 16)
#define TBLKS     21                   // ceil(1292/64)
#define FRAMES_PAD (TBLKS*FRAME_BLK)   // 1344
#define SIG_VALID (S_IN + NFFT)        // 663548 reflect-padded samples
#define SIG_STRIDE 689664              // (FRAMES_PAD-1)*HOP + NFFT, per (b,c) row

#define FTILES    72                   // freq padded to 1152 = 72*16 (>=1025, mult of 32)
#define PAIRS     36                   // 32-bin pairs (mel-GEMM K = 32); 9 per wave
#define KSTEPS    64                   // 2048 / 32

typedef __attribute__((ext_vector_type(16))) __bf16 v16bf;
typedef __attribute__((ext_vector_type(8)))  float  v8f;

__device__ __forceinline__ __bf16 to_bf16(float f) { return (__bf16)f; }

#define TWO_PI_OVER_N (6.283185307179586f / 2048.0f)

// ---------------- workspace layout (bytes) ----------------
// sig bf16:    [8][2][SIG_STRIDE]                       22,069,248 B
// basisRe bf16 swizzled: [FTILES][KSTEPS][32 lanes][8dw] 4,718,592 B
// basisIm bf16 swizzled:                                 4,718,592 B
// melPack bf16 swizzled: [PAIRS][8 ntile][32][8dw]         294,912 B
// total ~31.8 MB (assumes ws_size >= 32 MB)
#define SIG_BYTES   ((size_t)BATCH*NCH*SIG_STRIDE*2)
#define BASIS_BYTES ((size_t)FTILES*KSTEPS*32*8*4)
#define MELP_BYTES  ((size_t)PAIRS*8*32*8*4)
#define OFF_BRE     (SIG_BYTES)
#define OFF_BIM     (OFF_BRE + BASIS_BYTES)
#define OFF_MELP    (OFF_BIM + BASIS_BYTES)

// ---------------- kernel 1: windowed DFT basis, pre-swizzled for WMMA B ----
// B-matrix k order used here (contiguous, same as A; not AMD's table order): VGPR j, lane L:
//   K = 2j + (L>=16 ? 16 : 0) (+1 for high half), N = L%16
__global__ void build_basis_kernel(__bf16* __restrict__ bre,
                                   __bf16* __restrict__ bim) {
    int id = blockIdx.x * 256 + threadIdx.x;        // ((tile*64+ks)*32+lane)*8 + j
    int j    =  id        & 7;
    int lane = (id >> 3)  & 31;
    int ks   = (id >> 8)  & 63;
    int tile =  id >> 14;
    if (tile >= FTILES) return;
    int n = lane & 15;
    int f = tile * 16 + n;
    int kbase = ks * 32 + ((lane >> 4) << 4) + 2 * j;
    #pragma unroll
    for (int e = 0; e < 2; ++e) {
        int k = kbase + e;
        float re = 0.0f, im = 0.0f;
        if (f < NBINS) {
            float wdw = 0.5f - 0.5f * cosf((float)k * TWO_PI_OVER_N); // periodic Hann
            int ph = (k * f) & (NFFT - 1);                            // exact mod-N reduction
            float s, cn;
            sincosf((float)ph * TWO_PI_OVER_N, &s, &cn);
            re = wdw * cn;
            im = wdw * s;   // sign irrelevant: power = re^2 + im^2
        }
        bre[(size_t)id * 2 + e] = to_bf16(re);
        bim[(size_t)id * 2 + e] = to_bf16(im);
    }
}

// ---------------- kernel 2: mel filterbank, pre-swizzled for WMMA B --------
__global__ void build_melpack_kernel(const float* __restrict__ mel_fb,
                                     __bf16* __restrict__ melp) {
    int id = blockIdx.x * 256 + threadIdx.x;        // ((pair*8+nt)*32+lane)*8 + j
    int j    =  id        & 7;
    int lane = (id >> 3)  & 31;
    int nt   = (id >> 8)  & 7;
    int pair =  id >> 11;
    if (pair >= PAIRS) return;
    int melcol = nt * 16 + (lane & 15);
    int kbase  = 2 * j + ((lane >> 4) << 4);
    #pragma unroll
    for (int e = 0; e < 2; ++e) {
        int bin = pair * 32 + kbase + e;
        float v = (bin < NBINS) ? mel_fb[(size_t)melcol * NBINS + bin] : 0.0f;
        melp[(size_t)id * 2 + e] = to_bf16(v);
    }
}

// ---------------- kernel 3: reflect-pad + deinterleave + bf16 convert ------
__global__ void pack_signal_kernel(const float* __restrict__ x,
                                   __bf16* __restrict__ sig) {
    size_t id = (size_t)blockIdx.x * 256 + threadIdx.x;
    if (id >= (size_t)BATCH * NCH * SIG_STRIDE) return;
    int s  = (int)(id % SIG_STRIDE);
    int bc = (int)(id / SIG_STRIDE);
    int b = bc >> 1, c = bc & 1;
    float v = 0.0f;
    if (s < SIG_VALID) {
        int i = s - (NFFT / 2);
        if (i < 0)            i = -i;                 // reflect (no edge repeat)
        else if (i >= S_IN)   i = 2 * (S_IN - 1) - i;
        v = x[((size_t)b * S_IN + i) * NCH + c];
    }
    sig[id] = to_bf16(v);
}

// ---------------- kernel 4: fused STFT-power-mel via WMMA ------------------
// grid = (21, 2, 8) blocks of 128 threads (4 waves of 32).
// Each wave owns 16 frames; each wave streams its own basis pairs (pair = wid + 4*pi).
__global__ __launch_bounds__(128)
void stft_mel_kernel(const __bf16* __restrict__ sig,
                     const __bf16* __restrict__ bre,
                     const __bf16* __restrict__ bim,
                     const __bf16* __restrict__ melp,
                     float* __restrict__ out) {
    const int wid = threadIdx.x >> 5;
    const int L   = threadIdx.x & 31;
    const int tblk = blockIdx.x, c = blockIdx.y, b = blockIdx.z;
    const int tBase = tblk * FRAME_BLK + wid * 16;

    __shared__ float Pt[4][16][33];                 // per-wave 16x32 power, padded stride

    // A-matrix (16x32 bf16) per-lane addressing: M = L%16, contiguous run at
    // K = ko..ko+15, ko = (L>=16)*16
    const __bf16* arow = sig + (size_t)(b * NCH + c) * SIG_STRIDE
                             + (size_t)(tBase + (L & 15)) * HOP
                             + ((L >> 4) << 4);

    const float4* breq = (const float4*)bre;
    const float4* bimq = (const float4*)bim;
    const float4* mpq  = (const float4*)melp;

    v8f melAcc[8] = {};                             // 16 frames x 128 mels, f32

    for (int pi = 0; pi < 9; ++pi) {                // uniform across all 4 waves
        const int pair = wid + (pi << 2);
        const int tile0 = pair * 2, tile1 = tile0 + 1;
        v8f are0 = {}, aim0 = {}, are1 = {}, aim1 = {};

        for (int ks = 0; ks < KSTEPS; ++ks) {
            v16bf a;
            ((float4*)&a)[0] = *(const float4*)(arow + ks * 32);
            ((float4*)&a)[1] = *(const float4*)(arow + ks * 32 + 8);

            const size_t i0 = ((size_t)(tile0 * KSTEPS + ks) * 32 + L) * 2;
            const size_t i1 = ((size_t)(tile1 * KSTEPS + ks) * 32 + L) * 2;
            v16bf br0, bi0, br1, bi1;
            ((float4*)&br0)[0] = breq[i0];     ((float4*)&br0)[1] = breq[i0 + 1];
            ((float4*)&bi0)[0] = bimq[i0];     ((float4*)&bi0)[1] = bimq[i0 + 1];
            ((float4*)&br1)[0] = breq[i1];     ((float4*)&br1)[1] = breq[i1 + 1];
            ((float4*)&bi1)[0] = bimq[i1];     ((float4*)&bi1)[1] = bimq[i1 + 1];

            // pull next k-step's basis lines toward the WGP (global_prefetch_b8)
            __builtin_prefetch(breq + i0 + 64, 0, 3);
            __builtin_prefetch(bimq + i0 + 64, 0, 3);

            are0 = __builtin_amdgcn_wmma_f32_16x16x32_bf16(false, a, false, br0, (short)0, are0, false, false);
            aim0 = __builtin_amdgcn_wmma_f32_16x16x32_bf16(false, a, false, bi0, (short)0, aim0, false, false);
            are1 = __builtin_amdgcn_wmma_f32_16x16x32_bf16(false, a, false, br1, (short)0, are1, false, false);
            aim1 = __builtin_amdgcn_wmma_f32_16x16x32_bf16(false, a, false, bi1, (short)0, aim1, false, false);
        }

        // |STFT|^2 for this 16x32 bin pair (C/D layout: M = v + 8*(L>=16), N = L%16)
        v8f p0 = are0 * are0 + aim0 * aim0;
        v8f p1 = are1 * are1 + aim1 * aim1;
        {
            const int m0 = (L >> 4) << 3;
            const int n  = L & 15;
            #pragma unroll
            for (int v = 0; v < 8; ++v) {
                Pt[wid][m0 + v][n]      = p0[v];
                Pt[wid][m0 + v][n + 16] = p1[v];
            }
        }
        __syncthreads();

        // re-swizzle power tile into A-layout (16x32), quantize to bf16
        v16bf pa;
        {
            const int mr = L & 15;
            const int ko = (L >> 4) << 4;
            #pragma unroll
            for (int e = 0; e < 8; ++e) {
                pa[e]     = to_bf16(Pt[wid][mr][ko + e]);
                pa[e + 8] = to_bf16(Pt[wid][mr][ko + 8 + e]);
            }
        }

        // fused mel projection: K = 32 bins, 8 N-tiles of 16 mels
        #pragma unroll
        for (int nt = 0; nt < 8; ++nt) {
            const size_t mi = ((size_t)(pair * 8 + nt) * 32 + L) * 2;
            v16bf bm;
            ((float4*)&bm)[0] = mpq[mi];
            ((float4*)&bm)[1] = mpq[mi + 1];
            melAcc[nt] = __builtin_amdgcn_wmma_f32_16x16x32_bf16(false, pa, false, bm, (short)0, melAcc[nt], false, false);
        }
        __syncthreads();
    }

    // store: out[b][mel][t][c], exactly once per element (deterministic)
    const int m0 = (L >> 4) << 3;
    const int n  = L & 15;
    #pragma unroll
    for (int nt = 0; nt < 8; ++nt) {
        #pragma unroll
        for (int v = 0; v < 8; ++v) {
            const int t = tBase + m0 + v;
            if (t < NFRAMES) {
                const int mel = nt * 16 + n;
                out[(((size_t)b * NMELS + mel) * NFRAMES + t) * NCH + c] = melAcc[nt][v];
            }
        }
    }
}

// ---------------- launch ----------------
extern "C" void kernel_launch(void* const* d_in, const int* in_sizes, int n_in,
                              void* d_out, int out_size, void* d_ws, size_t ws_size,
                              hipStream_t stream) {
    const float* x      = (const float*)d_in[0];   // (8, 661500, 2) f32
    const float* mel_fb = (const float*)d_in[1];   // (128, 1025) f32
    float* out = (float*)d_out;                    // (8, 128, 1292, 2) f32

    char* w = (char*)d_ws;                         // needs ~31.8 MB
    __bf16* sig  = (__bf16*)(w);
    __bf16* bre  = (__bf16*)(w + OFF_BRE);
    __bf16* bim  = (__bf16*)(w + OFF_BIM);
    __bf16* melp = (__bf16*)(w + OFF_MELP);

    // 1) windowed DFT basis (bf16, WMMA-B swizzled): 72*64*32*8 dwords
    build_basis_kernel<<<(FTILES * KSTEPS * 32 * 8) / 256, 256, 0, stream>>>(bre, bim);
    // 2) mel filterbank pack: 36*8*32*8 dwords
    build_melpack_kernel<<<(PAIRS * 8 * 32 * 8) / 256, 256, 0, stream>>>(mel_fb, melp);
    // 3) reflect-pad + deinterleave + bf16 signal
    {
        size_t total = (size_t)BATCH * NCH * SIG_STRIDE;
        pack_signal_kernel<<<(unsigned)((total + 255) / 256), 256, 0, stream>>>(x, sig);
    }
    // 4) fused STFT-power-mel GEMM
    stft_mel_kernel<<<dim3(TBLKS, NCH, BATCH), 128, 0, stream>>>(sig, bre, bim, melp, out);
}
